// Head_75316546503257
// MI455X (gfx1250) — matrix operand order repaired and checked by reading, not compiled, not run on hardware
//
#include <hip/hip_runtime.h>
#include <hip/hip_bf16.h>
#include <stdint.h>

#define Bsz    4
#define Tsz    4096
#define DMODEL 1024
#define DKd    64

typedef __attribute__((ext_vector_type(16))) __bf16 v16bf;
typedef __attribute__((ext_vector_type(8)))  __bf16 v8bf;
typedef __attribute__((ext_vector_type(8)))  float  v8f;

union U16 { uint4 u; v8bf h; unsigned short s[8]; };

__device__ __forceinline__ __bf16 f2bf(float f) { return (__bf16)f; }
__device__ __forceinline__ unsigned short f2bfu(float f) {
  __bf16 b = (__bf16)f;
  return __builtin_bit_cast(unsigned short, b);
}

// 2 runs of 8 bf16 (p, p+16) -> one WMMA operand register group (2x b128 loads)
__device__ __forceinline__ v16bf load_bf_tile(const __bf16* p) {
  U16 lo, hi;
  lo.u = *(const uint4*)(p);
  hi.u = *(const uint4*)(p + 16);
  v16bf r;
  #pragma unroll
  for (int e = 0; e < 8; ++e) { r[e] = lo.h[e]; r[8 + e] = hi.h[e]; }
  return r;
}

// 16x32 f32 A-tile slice for one lane (2 runs of 8 floats) -> bf16 A layout
__device__ __forceinline__ v16bf load_x_tile(const float* p) {
  float4 a0 = *(const float4*)(p);
  float4 a1 = *(const float4*)(p + 4);
  float4 a2 = *(const float4*)(p + 16);
  float4 a3 = *(const float4*)(p + 20);
  v16bf A;
  A[0]=f2bf(a0.x); A[1]=f2bf(a0.y); A[2]=f2bf(a0.z); A[3]=f2bf(a0.w);
  A[4]=f2bf(a1.x); A[5]=f2bf(a1.y); A[6]=f2bf(a1.z); A[7]=f2bf(a1.w);
  A[8]=f2bf(a2.x); A[9]=f2bf(a2.y); A[10]=f2bf(a2.z); A[11]=f2bf(a2.w);
  A[12]=f2bf(a3.x); A[13]=f2bf(a3.y); A[14]=f2bf(a3.z); A[15]=f2bf(a3.w);
  return A;
}

// ---------------------------------------------------------------------------
// Kernel 0: pre-pack weights. Wt[mat][n][k] = bf16(W[k][n]), 64x1024 per mat.
// ---------------------------------------------------------------------------
__global__ __launch_bounds__(256)
void pack_w(const float* __restrict__ Wq, const float* __restrict__ Wk,
            const float* __restrict__ Wv, __bf16* __restrict__ wt)
{
  const int idx = blockIdx.x * 256 + threadIdx.x;  // 0 .. 3*64*1024-1
  const int mat = idx >> 16;
  const int r   = idx & 65535;
  const int n   = r >> 10;     // 0..63
  const int k   = r & 1023;    // 0..1023
  const float* W = (mat == 0) ? Wq : ((mat == 1) ? Wk : Wv);
  wt[(size_t)idx] = f2bf(W[(size_t)k * DKd + n]);
}

// ---------------------------------------------------------------------------
// Kernel 1: QKV projection (WMMA bf16, double-buffered k-loop) + RoPE on Q/K.
// One wave computes a 16-row x 64-col tile of one of {Q,K,V}.
// Q,K stored row-major bf16 [b*T+t][d]; V stored transposed bf16 Vt[b*64+d][t].
// ---------------------------------------------------------------------------
__global__ __launch_bounds__(256)
void qkv_rope(const float* __restrict__ x, const float* __restrict__ fc,
              const __bf16* __restrict__ wt,
              __bf16* __restrict__ qb, __bf16* __restrict__ kb,
              __bf16* __restrict__ vt)
{
  const int lane = threadIdx.x & 31;
  const int wid  = threadIdx.x >> 5;
  const int wg   = blockIdx.x * 8 + wid;        // 0 .. 3071
  const int mat  = wg % 3;                      // 0=Q 1=K 2=V
  const int rowtile = wg / 3;                   // 0 .. 1023
  const int row0 = rowtile << 4;
  const int lm = lane & 15;
  const int hl = lane >> 4;

  const float*  xa = x + (size_t)(row0 + lm) * DMODEL + hl * 8;
  const __bf16* wb = wt + ((size_t)mat << 16) + (size_t)hl * 8;

  v8f acc[4] = {v8f{}, v8f{}, v8f{}, v8f{}};
  v16bf A0, A1, BT0[4], BT1[4];

  // prologue: fill buffer 0 for k0 = 0
  A0 = load_x_tile(xa);
  #pragma unroll
  for (int nt = 0; nt < 4; ++nt)
    BT0[nt] = load_bf_tile(wb + (size_t)(nt * 16 + lm) * DMODEL);

  for (int k0 = 0; k0 < DMODEL; k0 += 64) {
    // prefetch k0+32 into buffer 1 (always in range: DMODEL % 64 == 0)
    A1 = load_x_tile(xa + k0 + 32);
    #pragma unroll
    for (int nt = 0; nt < 4; ++nt)
      BT1[nt] = load_bf_tile(wb + (size_t)(nt * 16 + lm) * DMODEL + k0 + 32);

    #pragma unroll
    for (int nt = 0; nt < 4; ++nt)
      acc[nt] = __builtin_amdgcn_wmma_f32_16x16x32_bf16(
          false, A0, false, BT0[nt], (short)0, acc[nt], false, false);

    // prefetch k0+64 into buffer 0
    if (k0 + 64 < DMODEL) {
      A0 = load_x_tile(xa + k0 + 64);
      #pragma unroll
      for (int nt = 0; nt < 4; ++nt)
        BT0[nt] = load_bf_tile(wb + (size_t)(nt * 16 + lm) * DMODEL + k0 + 64);
    }

    #pragma unroll
    for (int nt = 0; nt < 4; ++nt)
      acc[nt] = __builtin_amdgcn_wmma_f32_16x16x32_bf16(
          false, A1, false, BT1[nt], (short)0, acc[nt], false, false);
  }

  const int bidx = row0 >> 12;            // batch
  const int t0   = row0 & (Tsz - 1);      // position within batch

  if (mat <= 1) {
    __bf16* dst = (mat == 0) ? qb : kb;
    const float2* fcp = (const float2*)fc;   // [T][32] of (cos, sin)
    #pragma unroll
    for (int nt = 0; nt < 4; ++nt) {
      const int ncol = nt * 16 + lm;
      const int p = ncol >> 1;
      const float sgn = (ncol & 1) ? 1.0f : -1.0f;
      #pragma unroll
      for (int v = 0; v < 8; ++v) {
        float c = acc[nt][v];
        float other = __shfl_xor(c, 1, 32);  // pair partner (col n^1)
        int t = t0 + (hl << 3) + v;
        float2 f = fcp[t * 32 + p];
        float o = c * f.x + sgn * other * f.y;
        dst[(size_t)(row0 + (hl << 3) + v) * DKd + ncol] = f2bf(o);
      }
    }
  } else {
    // V: pack 8 contiguous-t bf16 per lane, store transposed Vt[b*64+d][t]
    #pragma unroll
    for (int nt = 0; nt < 4; ++nt) {
      const int ncol = nt * 16 + lm;
      U16 pk;
      #pragma unroll
      for (int v = 0; v < 8; ++v) pk.h[v] = f2bf(acc[nt][v]);
      *(uint4*)(vt + ((size_t)(bidx * DKd + ncol)) * Tsz + t0 + (hl << 3)) = pk.u;
    }
  }
}

__device__ __forceinline__ void load_k_tiles(const __bf16* kbase, int s0,
                                             int lm, int hl, v16bf KB[2][2]) {
  #pragma unroll
  for (int nt = 0; nt < 2; ++nt)
    #pragma unroll
    for (int kk = 0; kk < 2; ++kk)
      KB[nt][kk] = load_bf_tile(kbase + (size_t)(s0 + nt * 16 + lm) * DKd
                                      + kk * 32 + hl * 8);
}

// ---------------------------------------------------------------------------
// Kernel 2: flash attention, one wave per 16-query tile, 32-key blocks.
// K tiles prefetched one block ahead; V tiles loaded before the softmax/LDS
// section so their latency hides behind the exp/shuffle VALU work.
// ---------------------------------------------------------------------------
__global__ __launch_bounds__(256)
void attn(const __bf16* __restrict__ qb, const __bf16* __restrict__ kb,
          const __bf16* __restrict__ vt, float* __restrict__ out)
{
  __shared__ __align__(16) unsigned short pl[8][16 * 32];  // per-wave P tile
  const int lane = threadIdx.x & 31;
  const int wid  = threadIdx.x >> 5;
  const int wg   = blockIdx.x * 8 + wid;   // 0 .. 1023
  const int bidx = wg >> 8;                // 256 query tiles per batch
  const int q0   = (wg & 255) << 4;
  const int lm = lane & 15;
  const int hl = lane >> 4;

  const __bf16* kbase = kb + (size_t)bidx * Tsz * DKd;
  const __bf16* vbase = vt + (size_t)bidx * DKd * Tsz;

  // Q tile in A layout (loaded once): lane = row lm, K = g*16 + hl*8 + r
  v16bf QA[2];
  {
    const __bf16* qr = qb + ((size_t)bidx * Tsz + q0 + lm) * DKd + hl * 8;
    #pragma unroll
    for (int kk = 0; kk < 2; ++kk) {
      U16 lo, hi;
      lo.u = *(const uint4*)(qr + kk * 32);
      hi.u = *(const uint4*)(qr + kk * 32 + 16);
      #pragma unroll
      for (int e = 0; e < 8; ++e) { QA[kk][e] = lo.h[e]; QA[kk][8 + e] = hi.h[e]; }
    }
  }

  v8f O[4] = {v8f{}, v8f{}, v8f{}, v8f{}};
  float mrow[8], lrow[8];
  #pragma unroll
  for (int v = 0; v < 8; ++v) { mrow[v] = -1e30f; lrow[v] = 0.0f; }

  v16bf KB[2][2], VB[4];
  load_k_tiles(kbase, 0, lm, hl, KB);

  const int nblk = (q0 + 47) >> 5;   // key blocks covering s <= q0+15
  for (int j = 0; j < nblk; ++j) {
    const int s0 = j << 5;

    // S = Q * K^T  (two 16x16 f32 tiles over 32 keys)
    v8f S[2];
    #pragma unroll
    for (int nt = 0; nt < 2; ++nt) {
      v8f s = {};
      #pragma unroll
      for (int kk = 0; kk < 2; ++kk)
        s = __builtin_amdgcn_wmma_f32_16x16x32_bf16(
            false, QA[kk], false, KB[nt][kk], (short)0, s, false, false);
      S[nt] = s;
    }

    // issue V loads for this block early (consumed after softmax + LDS)
    #pragma unroll
    for (int nt = 0; nt < 4; ++nt)
      VB[nt] = load_bf_tile(vbase + (size_t)(nt * 16 + lm) * Tsz + s0 + hl * 8);

    // prefetch K tiles for the next block (safe to overwrite: register WAR)
    if (j + 1 < nblk) load_k_tiles(kbase, s0 + 32, lm, hl, KB);

    // scale + causal mask (only boundary block needs masking)
    const bool needmask = (s0 + 31 > q0);
    #pragma unroll
    for (int nt = 0; nt < 2; ++nt) {
      const int scol = s0 + nt * 16 + lm;
      #pragma unroll
      for (int v = 0; v < 8; ++v) {
        float val = S[nt][v] * 0.125f;                 // 1/sqrt(64)
        if (needmask && scol > q0 + (hl << 3) + v) val = -1e30f;
        S[nt][v] = val;
      }
    }

    // online softmax: row max / rescale / row sum (rows live in 16-lane halves)
    float alpha[8], mnew[8];
    #pragma unroll
    for (int v = 0; v < 8; ++v) {
      float r = fmaxf(S[0][v], S[1][v]);
      #pragma unroll
      for (int msk = 1; msk < 16; msk <<= 1) r = fmaxf(r, __shfl_xor(r, msk, 32));
      mnew[v] = fmaxf(mrow[v], r);
      alpha[v] = __expf(mrow[v] - mnew[v]);
      mrow[v] = mnew[v];
    }
    #pragma unroll
    for (int v = 0; v < 8; ++v) {
      float p0 = __expf(S[0][v] - mnew[v]);
      float p1 = __expf(S[1][v] - mnew[v]);
      S[0][v] = p0; S[1][v] = p1;
      float rs = p0 + p1;
      #pragma unroll
      for (int msk = 1; msk < 16; msk <<= 1) rs += __shfl_xor(rs, msk, 32);
      lrow[v] = lrow[v] * alpha[v] + rs;
    }

    // P (C layout) -> LDS row-major 16x32 bf16 -> reload in A layout
    #pragma unroll
    for (int nt = 0; nt < 2; ++nt)
      #pragma unroll
      for (int v = 0; v < 8; ++v)
        pl[wid][((hl << 3) + v) * 32 + nt * 16 + lm] = f2bfu(S[nt][v]);

    asm volatile("s_wait_dscnt 0" ::: "memory");  // same-wave LDS RAW

    v16bf PA;
    {
      const unsigned short* base = &pl[wid][lm * 32 + hl * 8];
      U16 lo, hi;
      lo.u = *(const uint4*)(base);
      hi.u = *(const uint4*)(base + 16);
      #pragma unroll
      for (int e = 0; e < 8; ++e) { PA[e] = lo.h[e]; PA[8 + e] = hi.h[e]; }
    }

    // O = alpha*O + P @ V
    #pragma unroll
    for (int nt = 0; nt < 4; ++nt) {
      #pragma unroll
      for (int v = 0; v < 8; ++v) O[nt][v] *= alpha[v];
      O[nt] = __builtin_amdgcn_wmma_f32_16x16x32_bf16(
          false, PA, false, VB[nt], (short)0, O[nt], false, false);
    }
  }

  // epilogue: divide by softmax sum, write f32 output
  #pragma unroll
  for (int nt = 0; nt < 4; ++nt) {
    #pragma unroll
    for (int v = 0; v < 8; ++v) {
      out[((size_t)bidx * Tsz + q0 + (hl << 3) + v) * DKd + nt * 16 + lm] =
          O[nt][v] / lrow[v];
    }
  }
}

// ---------------------------------------------------------------------------
extern "C" void kernel_launch(void* const* d_in, const int* in_sizes, int n_in,
                              void* d_out, int out_size, void* d_ws, size_t ws_size,
                              hipStream_t stream) {
  (void)in_sizes; (void)n_in; (void)out_size; (void)ws_size;
  const float* x  = (const float*)d_in[0];
  const float* fc = (const float*)d_in[1];
  const float* Wq = (const float*)d_in[2];
  const float* Wk = (const float*)d_in[3];
  const float* Wv = (const float*)d_in[4];
  float* out = (float*)d_out;

  const size_t R = (size_t)Bsz * Tsz;          // 16384 rows
  __bf16* qb = (__bf16*)d_ws;                  // 2 MB
  __bf16* kb = qb + R * DKd;                   // 2 MB
  __bf16* vt = kb + R * DKd;                   // 2 MB (transposed V)
  __bf16* wt = vt + R * DKd;                   // 384 KB (packed weights)

  // pre-pack weights: 3*64*1024 elements
  pack_w<<<768, 256, 0, stream>>>(Wq, Wk, Wv, wt);
  // 3072 waves, 8 waves/block
  qkv_rope<<<384, 256, 0, stream>>>(x, fc, wt, qb, kb, vt);
  // 1024 query tiles, 8 waves/block
  attn<<<128, 256, 0, stream>>>(qb, kb, vt, out);
}
